// AttentionOperation_32521492365427
// MI455X (gfx1250) — matrix operand order repaired and checked by reading, not compiled, not run on hardware
//
#include <hip/hip_runtime.h>
#include <math.h>

// ---------------------------------------------------------------------------
// CDNA5 / gfx1250 fused BN-attention-BN-GELU.
// wave32; matrix ops via v_wmma_f32_16x16x32_bf16.
// ---------------------------------------------------------------------------

typedef __attribute__((ext_vector_type(16))) __bf16 v16bf;
typedef __attribute__((ext_vector_type(8)))  __bf16 v8bf;
typedef __attribute__((ext_vector_type(4)))  __bf16 v4bf;
typedef __attribute__((ext_vector_type(8)))  float  v8f;

#define N_  8
#define H_  8
#define D_  64
#define C_  64
#define L_  1024
#define M_  1024
#define BH_ (N_ * H_)
#define BL  128       // L rows per block (8 waves x 16 rows)
#define BM  64        // M columns per inner tile
#define LT_ (L_ / BL) // 8 L-tiles
#define EPSQ 1e-3f

__device__ __forceinline__ v8f zero8() {
  v8f r;
#pragma unroll
  for (int i = 0; i < 8; ++i) r[i] = 0.0f;
  return r;
}

__device__ __forceinline__ v4bf pack4(float a, float b, float c, float d) {
  v4bf r;
  r[0] = (__bf16)a; r[1] = (__bf16)b; r[2] = (__bf16)c; r[3] = (__bf16)d;
  return r;
}

// Assemble a 16-element bf16 fragment from two contiguous 8-element (16B)
// LDS chunks — matches the ISA A/B VGPR layouts (elements 0..7 then 8..15).
__device__ __forceinline__ v16bf frag2(const __bf16* lo, const __bf16* hi) {
  v8bf a = *(const v8bf*)lo;
  v8bf b = *(const v8bf*)hi;
  v16bf r;
#pragma unroll
  for (int i = 0; i < 8; ++i) { r[i] = a[i]; r[i + 8] = b[i]; }
  return r;
}

// ---------------------------------------------------------------------------
// Kernel 1: per-block partial sum/sumsq of logits = Q^T K  (WMMA, bf16).
// BN mean & beta drop out of softmax (shift invariance); only var is needed.
// grid = (BH, LT_), block = 256 (8 waves).
// ---------------------------------------------------------------------------
__global__ __launch_bounds__(256)
void k_logit_stats(const float* __restrict__ q, const float* __restrict__ k,
                   float* __restrict__ psum, float* __restrict__ pssq) {
  __shared__ __align__(16) __bf16 Qt[BL][D_ + 8];  // [l][d], row 144B
  __shared__ __align__(16) __bf16 Kt[BM][D_ + 8];  // [m][d]
  __shared__ float red[8][2];

  const int bh = blockIdx.x, lt = blockIdx.y, l0 = lt * BL;
  const int tid = threadIdx.x;
  const int lane = tid & 31, wave = tid >> 5;
  const int ln = lane & 15, half = lane >> 4;

  // Stage Q tile: Qt[l][d] = q[bh][d][l0+l]; 4 d per thread -> b64 LDS stores
  for (int i = tid; i < BL * (D_ / 4); i += 256) {
    int l = i & (BL - 1), dq = i >> 7;
    const float* qp = &q[(bh * D_ + dq * 4) * L_ + l0 + l];
    *(v4bf*)&Qt[l][dq * 4] = pack4(qp[0], qp[L_], qp[2 * L_], qp[3 * L_]);
  }

  float sum = 0.f, ssq = 0.f;
  for (int m0 = 0; m0 < M_; m0 += BM) {
    __syncthreads();
    for (int i = tid; i < BM * (D_ / 4); i += 256) {
      int m = i & (BM - 1), dq = i >> 6;
      const float* kp = &k[(bh * D_ + dq * 4) * M_ + m0 + m];
      *(v4bf*)&Kt[m][dq * 4] = pack4(kp[0], kp[M_], kp[2 * M_], kp[3 * M_]);
    }
    __syncthreads();

    v8f acc[4];
#pragma unroll
    for (int t = 0; t < 4; ++t) acc[t] = zero8();
#pragma unroll
    for (int ks = 0; ks < 2; ++ks) {
      // A (16x32): row = l-row (lane&15); K chunks at half*8 and half*8+16
      const __bf16* ar = &Qt[wave * 16 + ln][ks * 32 + half * 8];
      v16bf a = frag2(ar, ar + 16);
#pragma unroll
      for (int t = 0; t < 4; ++t) {
        // B (32x16): col = m (lane&15); K chunks at half*8 and half*8+16
        const __bf16* br = &Kt[t * 16 + ln][ks * 32 + half * 8];
        v16bf b = frag2(br, br + 16);
        acc[t] = __builtin_amdgcn_wmma_f32_16x16x32_bf16(
            false, a, false, b, (short)0, acc[t], false, false);
      }
    }
#pragma unroll
    for (int t = 0; t < 4; ++t)
#pragma unroll
      for (int i = 0; i < 8; ++i) { float s = acc[t][i]; sum += s; ssq += s * s; }
  }

#pragma unroll
  for (int off = 16; off > 0; off >>= 1) {
    sum += __shfl_xor(sum, off, 32);
    ssq += __shfl_xor(ssq, off, 32);
  }
  if (lane == 0) { red[wave][0] = sum; red[wave][1] = ssq; }
  __syncthreads();
  if (tid == 0) {
    float s = 0.f, s2 = 0.f;
    for (int w = 0; w < 8; ++w) { s += red[w][0]; s2 += red[w][1]; }
    psum[bh * LT_ + lt] = s;
    pssq[bh * LT_ + lt] = s2;
  }
}

// ---------------------------------------------------------------------------
// Kernel 2: fold partials -> scale[h] = gamma_sim[h] * rsqrt(var_h + eps)
// ---------------------------------------------------------------------------
__global__ void k_head_scale(const float* __restrict__ psum,
                             const float* __restrict__ pssq,
                             const float* __restrict__ gamma_sim,
                             float* __restrict__ scale) {
  int h = threadIdx.x;  // 0..7
  float s = 0.f, s2 = 0.f;
  for (int b = 0; b < N_; ++b)
    for (int lt = 0; lt < LT_; ++lt) {
      int idx = (b * H_ + h) * LT_ + lt;
      s += psum[idx]; s2 += pssq[idx];
    }
  const float cnt = (float)N_ * (float)L_ * (float)M_;
  float mean = s / cnt;
  float var = s2 / cnt - mean * mean;
  scale[h] = gamma_sim[h] * rsqrtf(var + EPSQ);
}

// ---------------------------------------------------------------------------
// Kernel 3: flash attention with recomputed logits.
//   S = scale_h * (Q^T K) tile; online softmax; O += P * V^T, all WMMA bf16.
// grid = (BH, LT_), block = 256.
// ---------------------------------------------------------------------------
__global__ __launch_bounds__(256)
void k_attention(const float* __restrict__ q, const float* __restrict__ k,
                 const float* __restrict__ v, const float* __restrict__ scale,
                 float* __restrict__ rv) {
  __shared__ __align__(16) __bf16 Qt[BL][D_ + 8];
  __shared__ __align__(16) __bf16 Kt[BM][D_ + 8];
  __shared__ __align__(16) __bf16 Vt[C_][BM + 8];   // Vt[c][m] (contiguous K)
  __shared__ __align__(16) __bf16 Pw[8][16][BM];    // per-wave P staging
  __shared__ float Ot[C_][BL + 1];                  // coalesced epilogue

  const int bh = blockIdx.x, lt = blockIdx.y, l0 = lt * BL;
  const int h = bh & (H_ - 1);
  const int tid = threadIdx.x;
  const int lane = tid & 31, wave = tid >> 5;
  const int ln = lane & 15, half = lane >> 4;
  const float sc = scale[h];

  for (int i = tid; i < BL * (D_ / 4); i += 256) {
    int l = i & (BL - 1), dq = i >> 7;
    const float* qp = &q[(bh * D_ + dq * 4) * L_ + l0 + l];
    *(v4bf*)&Qt[l][dq * 4] = pack4(qp[0], qp[L_], qp[2 * L_], qp[3 * L_]);
  }

  float mrow[8], lrow[8];
  v8f o[4];
#pragma unroll
  for (int i = 0; i < 8; ++i) { mrow[i] = -3.0e38f; lrow[i] = 0.f; }
#pragma unroll
  for (int t = 0; t < 4; ++t) o[t] = zero8();

  for (int m0 = 0; m0 < M_; m0 += BM) {
    __syncthreads();
    for (int i = tid; i < BM * (D_ / 4); i += 256) {
      int m = i & (BM - 1), dq = i >> 6;
      const float* kp = &k[(bh * D_ + dq * 4) * M_ + m0 + m];
      *(v4bf*)&Kt[m][dq * 4] = pack4(kp[0], kp[M_], kp[2 * M_], kp[3 * M_]);
    }
    // Vt: m contiguous on both sides -> float4 loads + b64 packed stores
    for (int i = tid; i < C_ * (BM / 4); i += 256) {
      int mq = i & (BM / 4 - 1), c = i >> 4;
      float4 f = *(const float4*)&v[(bh * C_ + c) * M_ + m0 + mq * 4];
      *(v4bf*)&Vt[c][mq * 4] = pack4(f.x, f.y, f.z, f.w);
    }
    __syncthreads();

    // ---- S tile (16 rows x 64 cols) ----
    v8f acc[4];
#pragma unroll
    for (int t = 0; t < 4; ++t) acc[t] = zero8();
#pragma unroll
    for (int ks = 0; ks < 2; ++ks) {
      const __bf16* ar = &Qt[wave * 16 + ln][ks * 32 + half * 8];
      v16bf a = frag2(ar, ar + 16);
#pragma unroll
      for (int t = 0; t < 4; ++t) {
        const __bf16* br = &Kt[t * 16 + ln][ks * 32 + half * 8];
        v16bf b = frag2(br, br + 16);
        acc[t] = __builtin_amdgcn_wmma_f32_16x16x32_bf16(
            false, a, false, b, (short)0, acc[t], false, false);
      }
    }
#pragma unroll
    for (int t = 0; t < 4; ++t)
#pragma unroll
      for (int i = 0; i < 8; ++i) acc[t][i] *= sc;

    // ---- online softmax: C/D layout row = half*8+i, spread over 16 lanes ----
#pragma unroll
    for (int i = 0; i < 8; ++i) {
      float tm = acc[0][i];
#pragma unroll
      for (int t = 1; t < 4; ++t) tm = fmaxf(tm, acc[t][i]);
#pragma unroll
      for (int off = 8; off > 0; off >>= 1) tm = fmaxf(tm, __shfl_xor(tm, off, 32));
      float nm = fmaxf(mrow[i], tm);
      float corr = __expf(mrow[i] - nm);
      float ts = 0.f;
#pragma unroll
      for (int t = 0; t < 4; ++t) { float p = __expf(acc[t][i] - nm); acc[t][i] = p; ts += p; }
#pragma unroll
      for (int off = 8; off > 0; off >>= 1) ts += __shfl_xor(ts, off, 32);
      lrow[i] = lrow[i] * corr + ts;
      mrow[i] = nm;
#pragma unroll
      for (int t = 0; t < 4; ++t) o[t][i] *= corr;
    }

    // ---- P: C-layout -> A-layout via LDS (intra-wave) ----
#pragma unroll
    for (int t = 0; t < 4; ++t)
#pragma unroll
      for (int i = 0; i < 8; ++i)
        Pw[wave][half * 8 + i][t * 16 + ln] = (__bf16)acc[t][i];
    asm volatile("s_wait_dscnt 0x0" ::: "memory");  // cross-lane LDS hazard
    __builtin_amdgcn_wave_barrier();

    // ---- O += P (16x64) * V^T (64x16-tiles) ----
#pragma unroll
    for (int ks = 0; ks < 2; ++ks) {
      const __bf16* ar = &Pw[wave][ln][ks * 32 + half * 8];
      v16bf a = frag2(ar, ar + 16);
#pragma unroll
      for (int t = 0; t < 4; ++t) {
        const __bf16* br = &Vt[t * 16 + ln][ks * 32 + half * 8];
        v16bf b = frag2(br, br + 16);
        o[t] = __builtin_amdgcn_wmma_f32_16x16x32_bf16(
            false, a, false, b, (short)0, o[t], false, false);
      }
    }
  }

  // ---- epilogue: O / rowsum, stage to LDS, coalesced store of rv[bh][c][l] ----
  __syncthreads();
#pragma unroll
  for (int t = 0; t < 4; ++t)
#pragma unroll
    for (int i = 0; i < 8; ++i)
      Ot[t * 16 + ln][wave * 16 + half * 8 + i] = o[t][i] / lrow[i];
  __syncthreads();
  for (int i = tid; i < C_ * BL; i += 256) {
    int c = i >> 7, l = i & (BL - 1);
    rv[(bh * C_ + c) * L_ + l0 + l] = Ot[c][l];
  }
}

// ---------------------------------------------------------------------------
// Kernel 4: per-channel (H*C=512) sum/sumsq over (N,L). One block per channel.
// ---------------------------------------------------------------------------
__global__ __launch_bounds__(256)
void k_chan_stats(const float* __restrict__ rv,
                  float* __restrict__ csum, float* __restrict__ cssq) {
  __shared__ float red[8][2];
  const int ch = blockIdx.x;           // ch = h*C + c
  const int tid = threadIdx.x;
  const int lane = tid & 31, wave = tid >> 5;
  float s = 0.f, s2 = 0.f;
  for (int i = tid; i < N_ * (L_ / 4); i += 256) {
    int b = i >> 8, lq = i & (L_ / 4 - 1);
    float4 x = *(const float4*)&rv[(b * (H_ * C_) + ch) * L_ + lq * 4];
    s += x.x + x.y + x.z + x.w;
    s2 += x.x * x.x + x.y * x.y + x.z * x.z + x.w * x.w;
  }
#pragma unroll
  for (int off = 16; off > 0; off >>= 1) {
    s += __shfl_xor(s, off, 32);
    s2 += __shfl_xor(s2, off, 32);
  }
  if (lane == 0) { red[wave][0] = s; red[wave][1] = s2; }
  __syncthreads();
  if (tid == 0) {
    float a = 0.f, a2 = 0.f;
    for (int w = 0; w < 8; ++w) { a += red[w][0]; a2 += red[w][1]; }
    csum[ch] = a; cssq[ch] = a2;
  }
}

// ---------------------------------------------------------------------------
// Kernel 5: BatchNorm1d + exact GELU (erf). rv layout == out layout.
// float4 vectorized; L=1024 is 4-aligned so all 4 lanes share one channel.
// ---------------------------------------------------------------------------
__global__ __launch_bounds__(256)
void k_bn_gelu(const float* __restrict__ rv, const float* __restrict__ csum,
               const float* __restrict__ cssq,
               const float* __restrict__ gamma_val,
               const float* __restrict__ beta_val, float* __restrict__ out) {
  int i4 = blockIdx.x * 256 + threadIdx.x;
  if (i4 >= (N_ * H_ * C_ * L_) / 4) return;
  int idx = i4 * 4;
  int ch = (idx >> 10) & (H_ * C_ - 1);
  const float cnt = (float)(N_ * L_);
  float mean = csum[ch] / cnt;
  float var = cssq[ch] / cnt - mean * mean;
  float rstd = rsqrtf(var + EPSQ);
  float g = gamma_val[ch], be = beta_val[ch];
  float4 x = *(const float4*)&rv[idx];
  float4 r;
  float* xe = &x.x; float* re = &r.x;
#pragma unroll
  for (int j = 0; j < 4; ++j) {
    float y = (xe[j] - mean) * rstd * g + be;
    re[j] = 0.5f * y * (1.0f + erff(y * 0.70710678118654752f));
  }
  *(float4*)&out[idx] = r;
}

// ---------------------------------------------------------------------------
// Launcher. Inputs (setup_inputs order, all fp32):
//   0: query (N,H,D,L) 1: key (N,H,D,M) 2: value (N,H,C,M)
//   3: gamma_sim (H)   4: beta_sim (H)  [beta/mean drop out of softmax]
//   5: gamma_val (H*C) 6: beta_val (H*C)
// Workspace: rv 16MB + 2056 floats of stats. No atomics -> deterministic.
// ---------------------------------------------------------------------------
extern "C" void kernel_launch(void* const* d_in, const int* in_sizes, int n_in,
                              void* d_out, int out_size, void* d_ws, size_t ws_size,
                              hipStream_t stream) {
  const float* q         = (const float*)d_in[0];
  const float* k         = (const float*)d_in[1];
  const float* v         = (const float*)d_in[2];
  const float* gamma_sim = (const float*)d_in[3];
  const float* gamma_val = (const float*)d_in[5];
  const float* beta_val  = (const float*)d_in[6];
  float* out = (float*)d_out;

  const size_t RVN = (size_t)N_ * H_ * C_ * L_;  // 4M floats
  float* rv    = (float*)d_ws;
  float* psum  = rv + RVN;                        // BH_*LT_ = 512
  float* pssq  = psum + BH_ * LT_;                // 512
  float* scale = pssq + BH_ * LT_;                // 8
  float* csum  = scale + H_;                      // 512
  float* cssq  = csum + H_ * C_;                  // 512

  dim3 gAttn(BH_, LT_);
  k_logit_stats<<<gAttn, 256, 0, stream>>>(q, k, psum, pssq);
  k_head_scale<<<1, H_, 0, stream>>>(psum, pssq, gamma_sim, scale);
  k_attention<<<gAttn, 256, 0, stream>>>(q, k, v, scale, rv);
  k_chan_stats<<<H_ * C_, 256, 0, stream>>>(rv, csum, cssq);
  k_bn_gelu<<<(unsigned)((RVN / 4 + 255) / 256), 256, 0, stream>>>(
      rv, csum, cssq, gamma_val, beta_val, out);
}
